// MiniCPMFlashAttention2_34643206210204
// MI455X (gfx1250) — matrix operand order repaired and checked by reading, not compiled, not run on hardware
//
#include <hip/hip_runtime.h>

typedef __attribute__((ext_vector_type(16))) __bf16   v16bf;
typedef __attribute__((ext_vector_type(16))) _Float16 v16h;
typedef __attribute__((ext_vector_type(8)))  float    v8f;

#define S_LEN 4096
#define HQ    32
#define HK    2
#define G_    16
#define D_    128
#define C_    255
#define CP    256
#define NB_   64
#define TOPK_ 16
#define NEGF  (-1e30f)
#define BIGF  (1e30f)
#define SCALE 0.08838834764831845f  // 1/sqrt(128)

// ---------------------------------------------------------------------------
// Zero the chunk_score accumulator (ws is poisoned by the harness).
// ---------------------------------------------------------------------------
__global__ void nsa_zero(float* __restrict__ p, int n) {
  int i = blockIdx.x * blockDim.x + threadIdx.x;
  if (i < n) p[i] = 0.f;
}

// ---------------------------------------------------------------------------
// CompressKV: mean-pool 32-wide windows (stride 16) of kv (S,2,Hk,D).
// ckb  : bf16 [hk][c][d]   (row-major, K-pairs along d contiguous for QK^T B)
// cvbT : f16  [hk][d][c]   (transposed, K-pairs along c contiguous for PV B)
// c = 255 is zero padding.
// ---------------------------------------------------------------------------
__global__ __launch_bounds__(128) void nsa_compress(
    const float* __restrict__ kv, __bf16* __restrict__ ckb,
    _Float16* __restrict__ cvbT) {
  int d  = threadIdx.x;
  int c  = blockIdx.x & 255;
  int hk = blockIdx.x >> 8;
  if (c < C_) {
    float sk = 0.f, sv = 0.f;
    for (int i = 0; i < 32; ++i) {
      int s = c * 16 + i;
      sk += kv[((s * 2 + 0) * HK + hk) * D_ + d];
      sv += kv[((s * 2 + 1) * HK + hk) * D_ + d];
    }
    ckb [(hk * CP + c) * D_ + d] = (__bf16)(sk * (1.f / 32.f));
    cvbT[(hk * D_ + d) * CP + c] = (_Float16)(sv * (1.f / 32.f));
  } else {
    ckb [(hk * CP + c) * D_ + d] = (__bf16)0.f;
    cvbT[(hk * D_ + d) * CP + c] = (_Float16)0.f;
  }
}

// ---------------------------------------------------------------------------
// Compressed attention.
// 4 wave32's per workgroup; wave w handles GQA head g = gq*4 + w of the SAME
// (hk, 16-row s-tile) -> identical ck/cv B-fragment addresses across waves
// (served by WGP$), and the 4 waves' P tiles are reduced in LDS so only one
// set of chunk_score global atomics is issued per block.
// QK^T: bf16 WMMA. PV: f16 WMMA. f32 accumulate. Softmax in LDS (f32).
// ---------------------------------------------------------------------------
__global__ __launch_bounds__(128) void nsa_cmp_attn(
    const float* __restrict__ q, const __bf16* __restrict__ ckb,
    const _Float16* __restrict__ cvbT, float* __restrict__ out,
    float* __restrict__ chunk_score) {
  __shared__ float sc[4][16][CP];  // 4 x 16KB score tiles = 64 KB

  int tid  = threadIdx.x;
  int wave = tid >> 5;
  int lane = tid & 31;
  int half = lane >> 4, ln = lane & 15;

  int blk   = blockIdx.x;          // HK * 4 * 256 = 2048 blocks
  int stile = blk & 255;
  int gq    = (blk >> 8) & 3;
  int hk    = blk >> 10;
  int s0    = stile * 16;
  int g     = gq * 4 + wave;
  int h     = hk * G_ + g;         // q head index (reshape S,Hk,G,D)

  float (*myc)[CP] = sc[wave];

  // ---- Q A-fragments (ISA 16-bit A 16x32 layout): lane row = ln,
  // VGPR v<4: K = half*8 + 2v ; v>=4: K = 16 + half*8 + 2(v-4)
  const float* qrow = q + ((s0 + ln) * HQ + h) * D_;
  v16bf A[4];
#pragma unroll
  for (int kc = 0; kc < 4; ++kc) {
#pragma unroll
    for (int v = 0; v < 8; ++v) {
      int kb = ((v < 4) ? 0 : 16) + half * 8 + 2 * (v & 3);
      int d  = kc * 32 + kb;
      A[kc][2 * v]     = (__bf16)qrow[d];
      A[kc][2 * v + 1] = (__bf16)qrow[d + 1];
    }
  }

  // ---- logits = Q * ck^T, per 16-column c-tile
  const __bf16* ckh = ckb + hk * CP * D_;
  for (int ct = 0; ct < 16; ++ct) {
    v8f acc = {0.f, 0.f, 0.f, 0.f, 0.f, 0.f, 0.f, 0.f};
    int c = ct * 16 + ln;                      // B column N = ln
    const __bf16* krow = ckh + c * D_;
#pragma unroll
    for (int kc = 0; kc < 4; ++kc) {
      v16bf B;                                 // B 32x16: K = (v<4?0:16) + half*8 + 2(v&3)
#pragma unroll
      for (int v = 0; v < 8; ++v) {
        int d = kc * 32 + ((v < 4) ? 0 : 16) + half * 8 + 2 * (v & 3);
        B[2 * v]     = krow[d];
        B[2 * v + 1] = krow[d + 1];
      }
      acc = __builtin_amdgcn_wmma_f32_16x16x32_bf16(false, A[kc], false, B,
                                                    (short)0, acc, false, false);
    }
    // C/D layout: row m = half*8 + v, col = ln ; causal mask s >= 16c+31
#pragma unroll
    for (int v = 0; v < 8; ++v) {
      int m = half * 8 + v;
      int s = s0 + m;
      bool vis = (c < C_) && (s >= 16 * c + 31);
      myc[m][c] = vis ? acc[v] * SCALE : NEGF;
    }
  }
  __syncthreads();

  // ---- softmax per row (lanes 0..15 of each wave, one row each)
  if (lane < 16) {
    int s = s0 + lane;
    float* row = myc[lane];
    if (s < 31) {  // has_kv == false -> p = 0
      for (int c2 = 0; c2 < CP; ++c2) row[c2] = 0.f;
    } else {
      float mx = NEGF;
      for (int c2 = 0; c2 < CP; ++c2) mx = fmaxf(mx, row[c2]);
      float sum = 0.f;
      for (int c2 = 0; c2 < CP; ++c2) {
        float e = __expf(row[c2] - mx);
        row[c2] = e;
        sum += e;
      }
      float inv = 1.f / sum;
      for (int c2 = 0; c2 < CP; ++c2) row[c2] *= inv;
    }
  }
  __syncthreads();  // all 4 tiles final & read-only from here on

  // ---- chunk_score[hk][s][c] += sum over the block's 4 g heads
  // (plain LDS reads + one global atomic per element; 4x fewer atomics)
  {
    const float* t0 = &sc[0][0][0];
    const float* t1 = &sc[1][0][0];
    const float* t2 = &sc[2][0][0];
    const float* t3 = &sc[3][0][0];
    for (int i = tid; i < 16 * CP; i += 128) {
      float v4 = t0[i] + t1[i] + t2[i] + t3[i];
      int m = i >> 8, c2 = i & 255;
      atomicAdd(&chunk_score[(hk * S_LEN + s0 + m) * CP + c2], v4);
    }
  }

  // ---- out = P * cv  (f16 WMMA, K over c in chunks of 32)
  const _Float16* cvh = cvbT + hk * D_ * CP;
  for (int dt = 0; dt < 8; ++dt) {
    v8f acc = {0.f, 0.f, 0.f, 0.f, 0.f, 0.f, 0.f, 0.f};
    const _Float16* vrow = cvh + (dt * 16 + ln) * CP;  // B column N = d
#pragma unroll
    for (int ct = 0; ct < 8; ++ct) {
      v16h Ap, Bv;
#pragma unroll
      for (int v = 0; v < 8; ++v) {
        int kb = ((v < 4) ? 0 : 16) + half * 8 + 2 * (v & 3);
        int c2 = ct * 32 + kb;                 // A: row = ln, K = c
        Ap[2 * v]     = (_Float16)myc[ln][c2];
        Ap[2 * v + 1] = (_Float16)myc[ln][c2 + 1];
        int kB = c2;                           // B: K = c, same order as A
        Bv[2 * v]     = vrow[kB];
        Bv[2 * v + 1] = vrow[kB + 1];
      }
      acc = __builtin_amdgcn_wmma_f32_16x16x32_f16(false, Ap, false, Bv,
                                                   (short)0, acc, false, false);
    }
#pragma unroll
    for (int v = 0; v < 8; ++v) {
      int m = half * 8 + v;
      out[((s0 + m) * HQ + h) * D_ + dt * 16 + ln] = acc[v];
    }
  }
}

// ---------------------------------------------------------------------------
// Block scoring + masked stable top-16 + ascending sort + future->-1.
// One thread per (hk, s).
// ---------------------------------------------------------------------------
__global__ __launch_bounds__(256) void nsa_topk(
    const float* __restrict__ chunk_score, float* __restrict__ outIdx) {
  int t = blockIdx.x * blockDim.x + threadIdx.x;
  if (t >= HK * S_LEN) return;
  int hk = t / S_LEN;
  int s  = t % S_LEN;
  int qb = s / 64;
  const float* cs = chunk_score + (hk * S_LEN + s) * CP;

  float bs[NB_];
  for (int b = 0; b < NB_; ++b) {
    if (b > qb) { bs[b] = -BIGF; continue; }               // future
    if (b < 1 || b >= qb - 1) { bs[b] = BIGF; continue; }  // init | local
    // bscore[b] = sum of chunk scores c in [4b-1, 4b+3] (pad -> skip OOB)
    int lo = b * 4 - 1; if (lo < 0) lo = 0;
    int hi = b * 4 + 3; if (hi > C_ - 1) hi = C_ - 1;
    float sum = 0.f;
    for (int c = lo; c <= hi; ++c) sum += cs[c];
    bs[b] = sum;
  }

  // stable top-16 (earliest index wins ties, matching jax.lax.top_k)
  unsigned long long used = 0ull;
  int idx[TOPK_];
  for (int j = 0; j < TOPK_; ++j) {
    float best = -3.4e38f;
    int bi = 0;
    for (int b = 0; b < NB_; ++b) {
      if (!((used >> b) & 1ull) && bs[b] > best) { best = bs[b]; bi = b; }
    }
    used |= 1ull << bi;
    idx[j] = bi;
  }
  // ascending insertion sort
  for (int j = 1; j < TOPK_; ++j) {
    int key = idx[j], i2 = j - 1;
    while (i2 >= 0 && idx[i2] > key) { idx[i2 + 1] = idx[i2]; --i2; }
    idx[i2 + 1] = key;
  }
  for (int j = 0; j < TOPK_; ++j) {
    int v2 = (idx[j] >= qb) ? -1 : idx[j];
    outIdx[t * TOPK_ + j] = (float)v2;
  }
}

// ---------------------------------------------------------------------------
extern "C" void kernel_launch(void* const* d_in, const int* in_sizes, int n_in,
                              void* d_out, int out_size, void* d_ws,
                              size_t ws_size, hipStream_t stream) {
  const float* q  = (const float*)d_in[0];  // (S, Hq, D) f32
  const float* kv = (const float*)d_in[1];  // (S, 2, Hk, D) f32
  // d_in[2] = cu_seqlens (single sequence, unused)

  float* out    = (float*)d_out;                    // (S, Hq, D)
  float* outIdx = out + (size_t)S_LEN * HQ * D_;    // (Hk, S, 16) as floats

  // workspace layout
  char* ws = (char*)d_ws;
  __bf16*   ckb  = (__bf16*)ws;                                // HK*CP*D_
  _Float16* cvbT = (_Float16*)(ws + (size_t)HK * CP * D_ * 2); // HK*D_*CP
  float* chunk_score =
      (float*)(ws + 2 * (size_t)HK * CP * D_ * 2);             // HK*S*CP f32

  int ncs = HK * S_LEN * CP;
  nsa_zero<<<(ncs + 255) / 256, 256, 0, stream>>>(chunk_score, ncs);
  nsa_compress<<<HK * CP, 128, 0, stream>>>(kv, ckb, cvbT);
  nsa_cmp_attn<<<HK * 4 * (S_LEN / 16), 128, 0, stream>>>(q, ckb, cvbT, out,
                                                          chunk_score);
  nsa_topk<<<(HK * S_LEN + 255) / 256, 256, 0, stream>>>(chunk_score, outIdx);
}
